// DocRedModel_24936580120906
// MI455X (gfx1250) — matrix operand order repaired and checked by reading, compiled, not run on hardware
//
#include <hip/hip_runtime.h>
#include <hip/hip_bf16.h>
#include <math.h>

// ---------------- problem constants (match reference) ----------------
#define BATCH      4
#define SEQLEN     1024
#define NMENT      128
#define NPAIR      512
#define HID        1024
#define NHEADS     16
#define NENT       42
#define EMB        768
#define BLK        64
#define NCLS       97
#define NCLSPAD    128
#define PROJD      128
#define KCAT       2048              // 2*HID
#define KBIL       (EMB * BLK)       // 49152
#define NROWS      (BATCH * NPAIR)   // 2048

typedef __attribute__((ext_vector_type(16))) __bf16 v16bf;
typedef __attribute__((ext_vector_type(8)))  __bf16 v8bf;
typedef __attribute__((ext_vector_type(4)))  __bf16 v4bf;
typedef __attribute__((ext_vector_type(8)))  float  v8f;
typedef int v4i __attribute__((vector_size(16)));
typedef int v2i __attribute__((vector_size(8)));

__device__ __forceinline__ __bf16 f32_to_bf16(float f) {
    union { float f; unsigned u; } v; v.f = f;
    unsigned r = v.u + 0x7FFFu + ((v.u >> 16) & 1u);   // RNE
    unsigned short h = (unsigned short)(r >> 16);
    return __builtin_bit_cast(__bf16, h);
}
__device__ __forceinline__ float bf16_to_f32(__bf16 h) {
    unsigned u = ((unsigned)__builtin_bit_cast(unsigned short, h)) << 16;
    return __builtin_bit_cast(float, u);
}

// ---------------- CDNA5 async global->LDS copies (ASYNCcnt) ----------
__device__ __forceinline__ void cp_async_b128(void* lds, const void* g) {
#if __has_builtin(__builtin_amdgcn_global_load_async_to_lds_b128)
    __builtin_amdgcn_global_load_async_to_lds_b128(
        (__attribute__((address_space(1))) v4i*)g,
        (__attribute__((address_space(3))) v4i*)lds, 0, 0);
#else
    *(v8bf*)lds = *(const v8bf*)g;
#endif
}
__device__ __forceinline__ void cp_async_b64(void* lds, const void* g) {
#if __has_builtin(__builtin_amdgcn_global_load_async_to_lds_b64)
    __builtin_amdgcn_global_load_async_to_lds_b64(
        (__attribute__((address_space(1))) v2i*)g,
        (__attribute__((address_space(3))) v2i*)lds, 0, 0);
#else
    *(v4bf*)lds = *(const v4bf*)g;
#endif
}
template<int Nout>
__device__ __forceinline__ void wait_async() {
#if __has_builtin(__builtin_amdgcn_s_wait_asynccnt)
    __builtin_amdgcn_s_wait_asynccnt(Nout);
#else
    asm volatile("s_wait_asynccnt %0" :: "i"(Nout) : "memory");
#endif
}

// A fragment (16x32 bf16): lane half h needs K = {8h..8h+7} ++ {16+8h..+7}
__device__ __forceinline__ v16bf load_a_frag(const __bf16* rowp, int half) {
    v8bf lo = *(const v8bf*)(rowp + 8 * half);
    v8bf hi = *(const v8bf*)(rowp + 16 + 8 * half);
    return __builtin_shufflevector(lo, hi, 0,1,2,3,4,5,6,7,8,9,10,11,12,13,14,15);
}
// B fragment (32x16 bf16): lane half h, fixed column n, K = {8h..8h+7} ++ {16+8h..+7}
__device__ __forceinline__ v16bf load_b_frag(const __bf16* colp, int half) {
    v8bf lo = *(const v8bf*)(colp + 8 * half);
    v8bf hi = *(const v8bf*)(colp + 16 + 8 * half);
    return __builtin_shufflevector(lo, hi, 0,1,2,3,4,5,6,7,8,9,10,11,12,13,14,15);
}

// =====================================================================
// 1) Entity logsumexp pooling
// =====================================================================
__global__ __launch_bounds__(256) void pool_kernel(
    const float* __restrict__ ent_lhs, const int* __restrict__ labels,
    float* __restrict__ ent_emb)
{
    __shared__ int lab[NMENT];
    const int e = blockIdx.x, b = blockIdx.y, tid = threadIdx.x;
    if (tid < NMENT) lab[tid] = labels[b * NMENT + tid];
    __syncthreads();

    int cnt = 0;
    #pragma unroll 4
    for (int m = 0; m < NMENT; ++m) cnt += (lab[m] == e) ? 1 : 0;

    const float* base = ent_lhs + (long)b * NMENT * HID;
    #pragma unroll
    for (int j = 0; j < 4; ++j) {
        int d = tid + 256 * j;
        float mx = -3.0e38f;
        for (int m = 0; m < NMENT; ++m)
            if (lab[m] == e) { float v = base[(long)m * HID + d]; mx = fmaxf(mx, v); }
        float s = 0.0f;
        for (int m = 0; m < NMENT; ++m)
            if (lab[m] == e) s += expf(base[(long)m * HID + d] - mx);
        float out = (cnt > 0) ? (mx + logf(s)) : 0.0f;
        ent_emb[((long)(b * NENT + e)) * HID + d] = out;
    }
}

// =====================================================================
// 2) Attention scatter-mean
// =====================================================================
__global__ __launch_bounds__(256) void attn_pool_kernel(
    const float* __restrict__ attn, const int* __restrict__ labels,
    float* __restrict__ am)
{
    __shared__ int lab[NMENT];
    const int head = blockIdx.x, e = blockIdx.y, b = blockIdx.z, tid = threadIdx.x;
    if (tid < NMENT) lab[tid] = labels[b * NMENT + tid];
    __syncthreads();

    int cnt = 0;
    #pragma unroll 4
    for (int m = 0; m < NMENT; ++m) cnt += (lab[m] == e) ? 1 : 0;

    float acc[4] = {0.f, 0.f, 0.f, 0.f};
    const float* hb = attn + ((long)(b * NHEADS + head)) * NMENT * SEQLEN;
    for (int m = 0; m < NMENT; ++m) {
        if (lab[m] == e) {
            const float* p = hb + (long)m * SEQLEN;
            #pragma unroll
            for (int j = 0; j < 4; ++j) acc[j] += p[tid + 256 * j];
        }
    }
    float inv = 1.0f / (float)(cnt > 0 ? cnt : 1);
    float* dst = am + (((long)(b * NENT + e)) * NHEADS + head) * SEQLEN;
    #pragma unroll
    for (int j = 0; j < 4; ++j) dst[tid + 256 * j] = acc[j] * inv;
}

// =====================================================================
// 3) hta (normalized) -> bf16
// =====================================================================
__global__ __launch_bounds__(256) void hta_kernel(
    const float* __restrict__ am, const int* __restrict__ hts,
    __bf16* __restrict__ hta)
{
    __shared__ float red[256];
    const int r = blockIdx.x, b = blockIdx.y, tid = threadIdx.x;
    const int h0 = hts[((long)(b * NPAIR + r)) * 2 + 0];
    const int t0 = hts[((long)(b * NPAIR + r)) * 2 + 1];
    const float* ah = am + ((long)(b * NENT + h0)) * NHEADS * SEQLEN;
    const float* at = am + ((long)(b * NENT + t0)) * NHEADS * SEQLEN;

    float v[4]; float local = 0.0f;
    #pragma unroll
    for (int j = 0; j < 4; ++j) {
        int l = tid + 256 * j;
        float s = 0.0f;
        #pragma unroll
        for (int hd = 0; hd < NHEADS; ++hd)
            s += ah[(long)hd * SEQLEN + l] * at[(long)hd * SEQLEN + l];
        s *= (1.0f / (float)NHEADS);
        v[j] = s; local += s;
    }
    red[tid] = local; __syncthreads();
    for (int s = 128; s > 0; s >>= 1) {
        if (tid < s) red[tid] += red[tid + s];
        __syncthreads();
    }
    float scale = 1.0f / (red[0] + 1e-5f);
    __bf16* dst = hta + ((long)(b * NPAIR + r)) * SEQLEN;
    #pragma unroll
    for (int j = 0; j < 4; ++j) dst[tid + 256 * j] = f32_to_bf16(v[j] * scale);
}

// =====================================================================
// 4) concat -> bf16: catH[r,:] = [ent_emb[b,h0,:](f32->bf16), rel_bf[r,:]]
// =====================================================================
__global__ __launch_bounds__(256) void cat_kernel(
    const float* __restrict__ ent_emb, const __bf16* __restrict__ rel,
    const int* __restrict__ hts,
    __bf16* __restrict__ catH, __bf16* __restrict__ catT)
{
    const int r = blockIdx.x, tid = threadIdx.x;
    const int b = r >> 9, rr = r & (NPAIR - 1);
    const int h0 = hts[((long)(b * NPAIR + rr)) * 2 + 0];
    const int t0 = hts[((long)(b * NPAIR + rr)) * 2 + 1];
    const float* eh = ent_emb + ((long)(b * NENT + h0)) * HID;
    const float* et = ent_emb + ((long)(b * NENT + t0)) * HID;
    const __bf16* rl = rel + (long)r * HID;
    for (int k = tid; k < KCAT; k += 256) {
        __bf16 hv, tv;
        if (k < HID) { hv = f32_to_bf16(eh[k]); tv = f32_to_bf16(et[k]); }
        else         { __bf16 rv = rl[k - HID]; hv = rv; tv = rv; }
        catH[(long)r * KCAT + k] = hv;
        catT[(long)r * KCAT + k] = tv;
    }
}

// =====================================================================
// 5) Tiled transpose + f32->bf16 convert: out[n][k] = in[k][n]
//    in: K x N f32 (row stride N); out: NPAD x K bf16; rows n>=N zeroed.
//    grid (K/32, NPAD/32, Z), block (32, 8).
// =====================================================================
__global__ void transpose_conv_kernel(
    const float* __restrict__ in, __bf16* __restrict__ out,
    int K, int N, long strideIn, long strideOut)
{
    __shared__ float t[32][33];
    in  += (long)blockIdx.z * strideIn;
    out += (long)blockIdx.z * strideOut;
    const int k0 = blockIdx.x * 32, n0 = blockIdx.y * 32;
    #pragma unroll
    for (int j = 0; j < 4; ++j) {
        int k = k0 + threadIdx.y + j * 8, n = n0 + threadIdx.x;
        t[threadIdx.y + j * 8][threadIdx.x] = (n < N) ? in[(long)k * N + n] : 0.0f;
    }
    __syncthreads();
    #pragma unroll
    for (int j = 0; j < 4; ++j) {
        int n = n0 + threadIdx.y + j * 8, k = k0 + threadIdx.x;
        out[(long)n * K + k] = f32_to_bf16(t[threadIdx.x][threadIdx.y + j * 8]);
    }
}

// =====================================================================
// Generic batched GEMM: C = act(A @ BT^T + bias)
// A: M x K bf16 row-major. BT: NPAD x K bf16 (pre-transposed, padded).
// Block tile 32x128, 8 waves, wave = 16x32 strip (2 acc, shared A frag).
// Double-buffered LDS fed by async global->LDS copies (3 instrs/wave).
// =====================================================================
__global__ __launch_bounds__(256) void gemm_bf16_wmma(
    const __bf16* __restrict__ A, long lda, long strideA,
    const __bf16* __restrict__ BT, long ldbt, long strideB,
    void* __restrict__ Cv, long ldc, long strideC,
    int M, int N, int K, const float* __restrict__ bias, int act, int outbf)
{
    __shared__ __align__(16) __bf16 As[2][32][40];
    __shared__ __align__(16) __bf16 BsT[2][128][40];
    const int tid = threadIdx.x, z = blockIdx.z;
    A += (long)z * strideA; BT += (long)z * strideB;
    float*  Cf = (float*)Cv + (long)z * strideC;
    __bf16* Cb = (__bf16*)Cv + (long)z * strideC;
    const int bm0 = blockIdx.y * 32, bn0 = blockIdx.x * 128;
    const int wave = tid >> 5, lane = tid & 31;
    const int wm = (wave & 1) * 16, wn = (wave >> 1) * 32;
    const int half = lane >> 4, l16 = lane & 15;
    // A copy map: 32 rows x 64B, one b64 per thread
    const int ar = tid >> 3, ac8 = (tid & 7) * 4;           // 4 bf16 = 8B
    const __bf16* agp = A + (long)(bm0 + ar) * lda + ac8;
    // B copy map: 128 rows x 64B, two b128 per thread
    const int br = tid >> 1, bc = (tid & 1) * 16;           // 16 bf16 = 32B
    const __bf16* bgp = BT + (long)(bn0 + br) * ldbt + bc;

    v8f acc0 = {}, acc1 = {};
    // prologue: stage k=0 into buffer 0
    cp_async_b64(&As[0][ar][ac8], agp);
    cp_async_b128(&BsT[0][br][bc], bgp);
    cp_async_b128(&BsT[0][br][bc + 8], bgp + 8);
    int s = 0;
    for (int k0 = 0; k0 < K; k0 += 32) {
        const int k1 = k0 + 32;
        if (k1 < K) {   // prefetch next stage
            cp_async_b64(&As[s ^ 1][ar][ac8], agp + k1);
            cp_async_b128(&BsT[s ^ 1][br][bc], bgp + k1);
            cp_async_b128(&BsT[s ^ 1][br][bc + 8], bgp + k1 + 8);
            wait_async<3>();
        } else {
            wait_async<0>();
        }
        __syncthreads();
        v16bf af  = load_a_frag(&As[s][wm + l16][0], half);
        v16bf bf0 = load_b_frag(&BsT[s][wn + l16][0], half);
        v16bf bf1 = load_b_frag(&BsT[s][wn + 16 + l16][0], half);
        acc0 = __builtin_amdgcn_wmma_f32_16x16x32_bf16(
            false, af, false, bf0, (short)0, acc0, false, false);
        acc1 = __builtin_amdgcn_wmma_f32_16x16x32_bf16(
            false, af, false, bf1, (short)0, acc1, false, false);
        __syncthreads();
        s ^= 1;
    }
    #pragma unroll
    for (int i = 0; i < 8; ++i) {
        int gm = bm0 + wm + i + 8 * half;
        int n0 = bn0 + wn + l16, n1 = n0 + 16;
        float v0 = acc0[i], v1 = acc1[i];
        if (bias) {
            v0 += bias[n0 < N ? n0 : 0];
            v1 += bias[n1 < N ? n1 : 0];
        }
        if (act == 1) { v0 = tanhf(v0); v1 = tanhf(v1); }
        if (outbf) {
            if (n0 < N) Cb[(long)gm * ldc + n0] = f32_to_bf16(v0);
            if (n1 < N) Cb[(long)gm * ldc + n1] = f32_to_bf16(v1);
        } else {
            if (n0 < N) Cf[(long)gm * ldc + n0] = v0;
            if (n1 < N) Cf[(long)gm * ldc + n1] = v1;
        }
    }
}

// =====================================================================
// Bilinear GEMM: embeds = (h (x)_blk t) @ W_bil + b_bil, K = 49152.
// A synthesized from bf16 h,t; B = W_bilT (bf16, pre-transposed) via
// async double-buffer (2 instrs/wave). Dual store: f32 out + bf16 copy.
// =====================================================================
__global__ __launch_bounds__(256) void bilinear_gemm_wmma(
    const __bf16* __restrict__ Hm, const __bf16* __restrict__ Tm,
    const __bf16* __restrict__ WT, float* __restrict__ Cf,
    __bf16* __restrict__ Cb, const float* __restrict__ bias)
{
    __shared__ __align__(16) __bf16 As[2][32][40];
    __shared__ __align__(16) __bf16 BsT[2][128][40];
    const int tid = threadIdx.x;
    const int bm0 = blockIdx.y * 32, bn0 = blockIdx.x * 128;
    const int wave = tid >> 5, lane = tid & 31;
    const int wm = (wave & 1) * 16, wn = (wave >> 1) * 32;
    const int half = lane >> 4, l16 = lane & 15;
    const int arow = tid >> 3, ac4 = (tid & 7) * 4;
    const int br = tid >> 1, bc = (tid & 1) * 16;
    const __bf16* bgp = WT + (long)(bn0 + br) * (long)KBIL + bc;

    // A synthesis into stage buffer for K-chunk k0
    auto synthA = [&](int st, int k0) {
        const int nb = k0 >> 12, ci = (k0 >> 6) & 63, j0 = k0 & 63;
        int gm = bm0 + arow;
        float hv = bf16_to_f32(Hm[(long)gm * EMB + nb * BLK + ci]);
        v4bf tv = *(const v4bf*)&Tm[(long)gm * EMB + nb * BLK + j0 + ac4];
        v4bf pk;
        #pragma unroll
        for (int c = 0; c < 4; ++c) pk[c] = f32_to_bf16(hv * bf16_to_f32(tv[c]));
        *(v4bf*)&As[st][arow][ac4] = pk;
    };

    v8f acc0 = {}, acc1 = {};
    synthA(0, 0);
    cp_async_b128(&BsT[0][br][bc], bgp);
    cp_async_b128(&BsT[0][br][bc + 8], bgp + 8);
    int s = 0;
    for (int k0 = 0; k0 < KBIL; k0 += 32) {
        const int k1 = k0 + 32;
        if (k1 < KBIL) {
            synthA(s ^ 1, k1);
            cp_async_b128(&BsT[s ^ 1][br][bc], bgp + k1);
            cp_async_b128(&BsT[s ^ 1][br][bc + 8], bgp + k1 + 8);
            wait_async<2>();
        } else {
            wait_async<0>();
        }
        __syncthreads();
        v16bf af  = load_a_frag(&As[s][wm + l16][0], half);
        v16bf bf0 = load_b_frag(&BsT[s][wn + l16][0], half);
        v16bf bf1 = load_b_frag(&BsT[s][wn + 16 + l16][0], half);
        acc0 = __builtin_amdgcn_wmma_f32_16x16x32_bf16(
            false, af, false, bf0, (short)0, acc0, false, false);
        acc1 = __builtin_amdgcn_wmma_f32_16x16x32_bf16(
            false, af, false, bf1, (short)0, acc1, false, false);
        __syncthreads();
        s ^= 1;
    }
    #pragma unroll
    for (int i = 0; i < 8; ++i) {
        int gm = bm0 + wm + i + 8 * half;
        int n0 = bn0 + wn + l16, n1 = n0 + 16;
        float v0 = acc0[i] + bias[n0];
        float v1 = acc1[i] + bias[n1];
        Cf[(long)gm * EMB + n0] = v0;
        Cf[(long)gm * EMB + n1] = v1;
        Cb[(long)gm * EMB + n0] = f32_to_bf16(v0);
        Cb[(long)gm * EMB + n1] = f32_to_bf16(v1);
    }
}

// =====================================================================
// launcher
// =====================================================================
extern "C" void kernel_launch(void* const* d_in, const int* in_sizes, int n_in,
                              void* d_out, int out_size, void* d_ws, size_t ws_size,
                              hipStream_t stream) {
    const float* seq     = (const float*)d_in[0];
    const float* ent     = (const float*)d_in[1];
    const float* attn    = (const float*)d_in[2];
    const int*   labels  = (const int*)d_in[3];
    const int*   hts     = (const int*)d_in[4];
    const float* W_head  = (const float*)d_in[5];
    const float* b_head  = (const float*)d_in[6];
    const float* W_tail  = (const float*)d_in[7];
    const float* b_tail  = (const float*)d_in[8];
    const float* W_bil   = (const float*)d_in[9];
    const float* b_bil   = (const float*)d_in[10];
    const float* W_cls   = (const float*)d_in[11];
    const float* b_cls   = (const float*)d_in[12];
    const float* W_proj  = (const float*)d_in[13];
    const float* b_proj  = (const float*)d_in[14];

    // ---------------- workspace layout ----------------
    char* p = (char*)d_ws;
    float* ent_emb = (float*)p;  p += sizeof(float) * (long)BATCH * NENT * HID;
    float* am      = (float*)p;  p += sizeof(float) * (long)BATCH * NENT * NHEADS * SEQLEN;
    __bf16* hta    = (__bf16*)p; p += 2L * BATCH * NPAIR * SEQLEN;
    __bf16* rel    = (__bf16*)p; p += 2L * NROWS * HID;
    __bf16* catH   = (__bf16*)p; p += 2L * NROWS * KCAT;
    __bf16* catT   = (__bf16*)p; p += 2L * NROWS * KCAT;
    __bf16* Hm     = (__bf16*)p; p += 2L * NROWS * EMB;
    __bf16* Tm     = (__bf16*)p; p += 2L * NROWS * EMB;
    __bf16* emb_bf = (__bf16*)p; p += 2L * NROWS * EMB;
    __bf16* seqT   = (__bf16*)p; p += 2L * BATCH * HID * SEQLEN;
    __bf16* WTh    = (__bf16*)p; p += 2L * EMB * KCAT;
    __bf16* WTt    = (__bf16*)p; p += 2L * EMB * KCAT;
    __bf16* WTbil  = (__bf16*)p; p += 2L * EMB * (long)KBIL;
    __bf16* WTcls  = (__bf16*)p; p += 2L * NCLSPAD * EMB;
    __bf16* WTproj = (__bf16*)p; p += 2L * PROJD * EMB;

    float* embeds  = (float*)d_out;
    float* cls     = embeds + (long)NROWS * EMB;
    float* proj    = cls + (long)NROWS * NCLS;

    dim3 tb(32, 8);
    // one-time transpose+convert of all B operands (zero-padded N)
    transpose_conv_kernel<<<dim3(SEQLEN/32, HID/32, BATCH), tb, 0, stream>>>(
        seq, seqT, SEQLEN, HID, (long)SEQLEN*HID, (long)HID*SEQLEN);
    transpose_conv_kernel<<<dim3(KCAT/32, EMB/32, 1), tb, 0, stream>>>(
        W_head, WTh, KCAT, EMB, 0, 0);
    transpose_conv_kernel<<<dim3(KCAT/32, EMB/32, 1), tb, 0, stream>>>(
        W_tail, WTt, KCAT, EMB, 0, 0);
    transpose_conv_kernel<<<dim3(KBIL/32, EMB/32, 1), tb, 0, stream>>>(
        W_bil, WTbil, KBIL, EMB, 0, 0);
    transpose_conv_kernel<<<dim3(EMB/32, NCLSPAD/32, 1), tb, 0, stream>>>(
        W_cls, WTcls, EMB, NCLS, 0, 0);
    transpose_conv_kernel<<<dim3(EMB/32, PROJD/32, 1), tb, 0, stream>>>(
        W_proj, WTproj, EMB, PROJD, 0, 0);

    // pooling chain
    pool_kernel<<<dim3(NENT, BATCH), 256, 0, stream>>>(ent, labels, ent_emb);
    attn_pool_kernel<<<dim3(NHEADS, NENT, BATCH), 256, 0, stream>>>(attn, labels, am);
    hta_kernel<<<dim3(NPAIR, BATCH), 256, 0, stream>>>(am, hts, hta);

    // rel = hta @ seq  (bf16 out, batched over docs)
    gemm_bf16_wmma<<<dim3(HID/128, NPAIR/32, BATCH), 256, 0, stream>>>(
        hta, SEQLEN, (long)NPAIR * SEQLEN,
        seqT, SEQLEN, (long)HID * SEQLEN,
        rel, HID, (long)NPAIR * HID,
        NPAIR, HID, SEQLEN, nullptr, 0, 1);

    cat_kernel<<<dim3(NROWS), 256, 0, stream>>>(ent_emb, rel, hts, catH, catT);

    // h / t  (tanh, bf16 out)
    gemm_bf16_wmma<<<dim3(EMB/128, NROWS/32, 1), 256, 0, stream>>>(
        catH, KCAT, 0, WTh, KCAT, 0, Hm, EMB, 0,
        NROWS, EMB, KCAT, b_head, 1, 1);
    gemm_bf16_wmma<<<dim3(EMB/128, NROWS/32, 1), 256, 0, stream>>>(
        catT, KCAT, 0, WTt, KCAT, 0, Tm, EMB, 0,
        NROWS, EMB, KCAT, b_tail, 1, 1);

    // embeds (f32 out + bf16 copy)
    bilinear_gemm_wmma<<<dim3(EMB/128, NROWS/32, 1), 256, 0, stream>>>(
        Hm, Tm, WTbil, embeds, emb_bf, b_bil);

    // heads
    gemm_bf16_wmma<<<dim3(1, NROWS/32, 1), 256, 0, stream>>>(
        emb_bf, EMB, 0, WTcls, EMB, 0, cls, NCLS, 0,
        NROWS, NCLS, EMB, b_cls, 0, 0);
    gemm_bf16_wmma<<<dim3(1, NROWS/32, 1), 256, 0, stream>>>(
        emb_bf, EMB, 0, WTproj, EMB, 0, proj, PROJD, 0,
        NROWS, PROJD, EMB, b_proj, 1, 0);
}
